// CausalAttention_49813030699089
// MI455X (gfx1250) — matrix operand order repaired and checked by reading, not compiled, not run on hardware
//
#include <hip/hip_runtime.h>
#include <hip/hip_bf16.h>

typedef _Float16 v16h __attribute__((ext_vector_type(16)));
typedef _Float16 v8h  __attribute__((ext_vector_type(8)));
typedef float    v8f  __attribute__((ext_vector_type(8)));
typedef float    v4f  __attribute__((ext_vector_type(4)));

union U16 { v16h v; v8h h[2]; };

static constexpr int BB = 4;
static constexpr int SS = 2048;
static constexpr int DD = 1024;
static constexpr size_t T_BYTES = (size_t)BB * SS * DD * 2;  // 16 MB per f16 activation tensor
static constexpr size_t W_BYTES = (size_t)DD * DD * 2;       // 2 MB per f16 weight

// ws layout (bytes):
//   Qh : 0            (16 MB)  row-major [B*S, D] f16
//   Kh : T_BYTES      (16 MB)  row-major [B*S, D] f16
//   Vt : 2*T_BYTES    (16 MB)  transposed [B, D, S] f16
//   Xh : 3*T_BYTES    (16 MB)  row-major [B*S, D] f16
//   Wh : 4*T_BYTES    (3 x 2 MB) f16 copies of Wq, Wk, Wv

// ---------------------------------------------------------------------------
// Kernel 0: fp32 -> f16 elementwise convert (8 elements/thread, b128 in, b128 out)
// ---------------------------------------------------------------------------
__global__ __launch_bounds__(256)
void cvt_f32_to_f16_kernel(const float* __restrict__ in, _Float16* __restrict__ out)
{
    const size_t base = ((size_t)blockIdx.x * 256 + threadIdx.x) * 8;
    v4f x0 = *(const v4f*)(in + base);
    v4f x1 = *(const v4f*)(in + base + 4);
    v8h o;
    #pragma unroll
    for (int i = 0; i < 4; ++i) { o[i] = (_Float16)x0[i]; o[4 + i] = (_Float16)x1[i]; }
    *(v8h*)(out + base) = o;
}

// ---------------------------------------------------------------------------
// Kernel 1: QKV projection.  out = X @ W^T  (nn.Linear, bias-free)
// One wave computes a 32(M) x 64(N) tile: 2 A-frags x 4 B-frags = 8 WMMAs / k-chunk.
// All operands are pre-converted f16; B-fragments are single contiguous v16h loads.
// Q,K stored row-major f16; V stored transposed (Vt[b][e][s]) for the PV wmma.
// ---------------------------------------------------------------------------
__global__ __launch_bounds__(256)
void qkv_proj_kernel(const _Float16* __restrict__ Xh,
                     const _Float16* __restrict__ Wh,   // 3 weights back-to-back
                     _Float16* __restrict__ Qh,
                     _Float16* __restrict__ Kh,
                     _Float16* __restrict__ Vt)
{
    const int lane = threadIdx.x & 31;
    const int wave = blockIdx.x * 8 + (threadIdx.x >> 5);
    const int M_TILES = (BB * SS) / 32;   // 256
    const int N_TILES = DD / 64;          // 16
    const int p     = wave / (M_TILES * N_TILES);
    const int rem   = wave % (M_TILES * N_TILES);
    const int mtile = rem / N_TILES;
    const int ntile = rem % N_TILES;
    if (p >= 3) return;

    const _Float16* W = Wh + (size_t)p * DD * DD;

    const int m0 = mtile * 32;
    const int n0 = ntile * 64;
    const int hi = (lane >= 16) ? 1 : 0;
    const int lm = lane & 15;

    v8f acc[2][4];
    #pragma unroll
    for (int mt = 0; mt < 2; ++mt)
        #pragma unroll
        for (int e = 0; e < 4; ++e) acc[mt][e] = {};

    const _Float16* arow0 = Xh + (size_t)(m0 + lm) * DD;
    const _Float16* arow1 = Xh + (size_t)(m0 + 16 + lm) * DD;

    for (int kc = 0; kc < DD / 32; ++kc) {
        const int k0 = kc * 32;
        // A frags: lane M = lane%16; 16 contiguous K k0+(hi?16:0)..+15
        const int ka = k0 + (hi ? 16 : 0);
        U16 a0, a1;
        a0.h[0] = *(const v8h*)(arow0 + ka);
        a0.h[1] = *(const v8h*)(arow0 + ka + 8);
        a1.h[0] = *(const v8h*)(arow1 + ka);
        a1.h[1] = *(const v8h*)(arow1 + ka + 8);
        #pragma unroll
        for (int e = 0; e < 4; ++e) {
            // B frag: lane N = lane%16; 16 contiguous K per lane, split by lane half.
            // B[k][n] = W[n][k] -> column n of B is row n of W (contiguous 32 B).
            const int n = n0 + e * 16 + lm;
            v16h bf = *(const v16h*)(W + (size_t)n * DD + k0 + (hi ? 16 : 0));
            acc[0][e] = __builtin_amdgcn_wmma_f32_16x16x32_f16(
                false, a0.v, false, bf, (short)0, acc[0][e], false, false);
            acc[1][e] = __builtin_amdgcn_wmma_f32_16x16x32_f16(
                false, a1.v, false, bf, (short)0, acc[1][e], false, false);
        }
    }

    // C/D fragment: element (M = r + 8*hi, N = lane%16)
    for (int mt = 0; mt < 2; ++mt) {
        for (int e = 0; e < 4; ++e) {
            #pragma unroll
            for (int r = 0; r < 8; ++r) {
                const int m = m0 + mt * 16 + r + 8 * hi;
                const int n = n0 + e * 16 + lm;
                const _Float16 h = (_Float16)acc[mt][e][r];
                if (p == 0) {
                    Qh[(size_t)m * DD + n] = h;
                } else if (p == 1) {
                    Kh[(size_t)m * DD + n] = h;
                } else {
                    const int b = m / SS, s = m % SS;
                    Vt[((size_t)b * DD + n) * SS + s] = h;
                }
            }
        }
    }
}

// ---------------------------------------------------------------------------
// Kernel 2: causal flash attention.  One wave per (batch, 16-query tile).
// O accumulator (16 x 1024 fp32 = 64 KB) lives in LDS (320 KB/WGP on CDNA5).
// Per 32-key tile: 64 QK^T wmmas -> online softmax (shfl_xor width-16, v_exp)
// -> P staged via LDS (D-frag -> A-frag relayout) -> 64 PV wmmas with alpha
// rescale on the LDS-resident O frags.
// ---------------------------------------------------------------------------
__global__ __launch_bounds__(32)
void attn_kernel(const _Float16* __restrict__ Qh,
                 const _Float16* __restrict__ Kh,
                 const _Float16* __restrict__ Vt,
                 float* __restrict__ Out)
{
    extern __shared__ char smem[];
    float*    Olds = (float*)smem;                       // 64 echunks * 32 lanes * 8 f32 = 64 KB
    _Float16* Plds = (_Float16*)(smem + 64 * 1024);      // 16 x 32 halves = 1 KB

    const int lane = threadIdx.x & 31;
    const int hi   = (lane >= 16) ? 1 : 0;
    const int lm   = lane & 15;

    const int QT = SS / 16;                 // 128 query tiles
    const int b  = blockIdx.x / QT;
    const int qt = blockIdx.x % QT;
    const int q0 = qt * 16;

    const _Float16* Qb = Qh + (size_t)b * SS * DD;
    const _Float16* Kb = Kh + (size_t)b * SS * DD;
    const _Float16* Vb = Vt + (size_t)b * DD * SS;

    // zero the O accumulator
    for (int e = 0; e < 64; ++e) {
        v8f z = {};
        *(v8f*)(Olds + e * 256 + lane * 8) = z;
    }

    float mrow[8], lrow[8], alpha[8];
    #pragma unroll
    for (int r = 0; r < 8; ++r) { mrow[r] = -__builtin_inff(); lrow[r] = 0.0f; }

    const float scale = 0.03125f;           // 1/sqrt(1024)

    const int ktmax = (q0 + 15) / 32;
    for (int kt = 0; kt <= ktmax; ++kt) {
        const int k0 = kt * 32;

        // ---- scores: S = Q K^T, 16 x 32 tile (two n-half fragments) ----
        v8f c0 = {}, c1 = {};
        const _Float16* qrow  = Qb + (size_t)(q0 + lm) * DD;
        const _Float16* k0row = Kb + (size_t)(k0 + lm) * DD;        // keys k0..k0+15
        const _Float16* k1row = Kb + (size_t)(k0 + 16 + lm) * DD;   // keys k0+16..k0+31

        // prefetch next key tile into the cache hierarchy
        if (kt < ktmax) __builtin_prefetch(Kb + (size_t)(k0 + 32 + lm) * DD, 0, 1);

        for (int dc = 0; dc < DD / 32; ++dc) {
            const int d0 = dc * 32;
            const int da = d0 + (hi ? 16 : 0);
            U16 af;
            af.h[0] = *(const v8h*)(qrow + da);
            af.h[1] = *(const v8h*)(qrow + da + 8);

            const int db = d0 + (hi ? 16 : 0);
            v16h bf0 = *(const v16h*)(k0row + db);   // contiguous 32 B per lane
            c0 = __builtin_amdgcn_wmma_f32_16x16x32_f16(
                false, af.v, false, bf0, (short)0, c0, false, false);
            v16h bf1 = *(const v16h*)(k1row + db);
            c1 = __builtin_amdgcn_wmma_f32_16x16x32_f16(
                false, af.v, false, bf1, (short)0, c1, false, false);
        }

        // ---- online softmax (row = r + 8*hi, col = lane%16 within frag) ----
        #pragma unroll
        for (int r = 0; r < 8; ++r) {
            const int qi = q0 + r + 8 * hi;
            const int j0 = k0 + lm;
            const int j1 = k0 + 16 + lm;
            float t0 = (j0 <= qi) ? c0[r] * scale : -__builtin_inff();
            float t1 = (j1 <= qi) ? c1[r] * scale : -__builtin_inff();
            float tm = fmaxf(t0, t1);
            for (int off = 8; off >= 1; off >>= 1)
                tm = fmaxf(tm, __shfl_xor(tm, off, 16));
            const float mnew = fmaxf(mrow[r], tm);
            alpha[r] = __expf(mrow[r] - mnew);
            const float p0 = __expf(t0 - mnew);
            const float p1 = __expf(t1 - mnew);
            float rs = p0 + p1;
            for (int off = 8; off >= 1; off >>= 1)
                rs += __shfl_xor(rs, off, 16);
            lrow[r] = lrow[r] * alpha[r] + rs;
            mrow[r] = mnew;
            // P tile row-major 16 x 32 in LDS
            const int mloc = r + 8 * hi;
            Plds[mloc * 32 + lm]      = (_Float16)p0;
            Plds[mloc * 32 + 16 + lm] = (_Float16)p1;
        }
        asm volatile("s_wait_dscnt 0x0" ::: "memory");

        // ---- P -> A-fragment relayout + PV accumulation over all 64 e-chunks ----
        {
            const int pbase = lm * 32 + (hi ? 16 : 0);
            U16 pf;
            pf.h[0] = *(const v8h*)(Plds + pbase);
            pf.h[1] = *(const v8h*)(Plds + pbase + 8);

            for (int e = 0; e < 64; ++e) {
                // B frag from Vt: lane column e*16+lm, contiguous keys k0+(hi?16:0)..+15
                v16h vf = *(const v16h*)(Vb + (size_t)(e * 16 + lm) * SS + k0 + (hi ? 16 : 0));

                v8f oc = *(v8f*)(Olds + e * 256 + lane * 8);
                #pragma unroll
                for (int r = 0; r < 8; ++r) oc[r] *= alpha[r];
                oc = __builtin_amdgcn_wmma_f32_16x16x32_f16(
                    false, pf.v, false, vf, (short)0, oc, false, false);
                *(v8f*)(Olds + e * 256 + lane * 8) = oc;
            }
        }
    }

    // ---- epilogue: normalize by l and write fp32 output ----
    float inv[8];
    #pragma unroll
    for (int r = 0; r < 8; ++r) inv[r] = 1.0f / lrow[r];
    for (int e = 0; e < 64; ++e) {
        v8f oc = *(v8f*)(Olds + e * 256 + lane * 8);
        #pragma unroll
        for (int r = 0; r < 8; ++r) {
            const int m = q0 + r + 8 * hi;
            const int n = e * 16 + lm;
            Out[((size_t)b * SS + m) * DD + n] = oc[r] * inv[r];
        }
    }
}

// ---------------------------------------------------------------------------
extern "C" void kernel_launch(void* const* d_in, const int* in_sizes, int n_in,
                              void* d_out, int out_size, void* d_ws, size_t ws_size,
                              hipStream_t stream) {
    (void)in_sizes; (void)n_in; (void)out_size; (void)ws_size;
    const float* X  = (const float*)d_in[0];
    const float* Wq = (const float*)d_in[1];
    const float* Wk = (const float*)d_in[2];
    const float* Wv = (const float*)d_in[3];
    float* Out = (float*)d_out;

    char* ws = (char*)d_ws;
    _Float16* Qh = (_Float16*)(ws);
    _Float16* Kh = (_Float16*)(ws + T_BYTES);
    _Float16* Vt = (_Float16*)(ws + 2 * T_BYTES);
    _Float16* Xh = (_Float16*)(ws + 3 * T_BYTES);
    _Float16* Wh = (_Float16*)(ws + 4 * T_BYTES);   // 3 weights back-to-back

    // ---- convert fp32 inputs to f16 once (bandwidth-trivial) ----
    {
        const int xblocks = (BB * SS * DD) / (256 * 8);   // 4096
        hipLaunchKernelGGL(cvt_f32_to_f16_kernel, dim3(xblocks), dim3(256), 0, stream, X, Xh);
        const int wblocks = (DD * DD) / (256 * 8);        // 512
        hipLaunchKernelGGL(cvt_f32_to_f16_kernel, dim3(wblocks), dim3(256), 0, stream,
                           Wq, Wh);
        hipLaunchKernelGGL(cvt_f32_to_f16_kernel, dim3(wblocks), dim3(256), 0, stream,
                           Wk, Wh + (size_t)DD * DD);
        hipLaunchKernelGGL(cvt_f32_to_f16_kernel, dim3(wblocks), dim3(256), 0, stream,
                           Wv, Wh + 2 * (size_t)DD * DD);
    }

    // ---- projections: 3 proj * 256 mtiles * 16 ntiles = 12288 waves, 8 waves/block ----
    const int waves  = 3 * (BB * SS / 32) * (DD / 64);
    const int blocks = waves / 8;   // 1536
    hipLaunchKernelGGL(qkv_proj_kernel, dim3(blocks), dim3(256), 0, stream,
                       Xh, Wh, Qh, Kh, Vt);

    // ---- attention: one wave per (b, 16-query tile); 65 KB dynamic LDS per block ----
    const int ablocks = BB * (SS / 16);   // 512
    const size_t shmem = 64 * 1024 + 1024;
    hipLaunchKernelGGL(attn_kernel, dim3(ablocks), dim3(32), shmem, stream,
                       Qh, Kh, Vt, Out);
}
